// GCN_69458211110958
// MI455X (gfx1250) — matrix operand order repaired and checked by reading, not compiled, not run on hardware
//
#include <hip/hip_runtime.h>
#include <hip/hip_bf16.h>
#include <math.h>

// ---------------------------------------------------------------------------
// GCN pipeline for MI455X (gfx1250, wave32, WMMA):
//   H1 = X @ W1                      (bf16 WMMA, f32 accum, LDS-staged A)
//   A1 = leaky_relu(spmm(H1))        (edge scatter-add, f32 atomics)
//   H2 = A1 @ W2                     (bf16 WMMA)
//   X3 = spmm(H2)                    -> d_out[N*64 ..]
//   Y  = sigmoid(X3 @ W3)            -> d_out[0 .. N*64)
//
// GEMM: one block covers a 64-row x Nout strip. The 64x32 bf16 A tile is
// cooperatively staged into LDS (double-buffered) once per K-step and shared
// by all waves (one wave per 16-col N-tile). Templated on NTILES so the
// staging loop fully unrolls and all four A fragments are fetched before the
// WMMA chain (single dscnt wait, 4 back-to-back v_wmma).
// ---------------------------------------------------------------------------

typedef __attribute__((ext_vector_type(16))) __bf16 v16bf;
typedef __attribute__((ext_vector_type(8)))  __bf16 v8bf;
typedef __attribute__((ext_vector_type(8)))  float  v8f;

union AFrag { v16bf v; v8bf h[2]; };

#define NFEAT 256
#define NHID  128
#define NCLASS 64
#define LEAKY 0.01f

// ---- elementwise f32 -> bf16 convert, optional leaky_relu -------------------
__global__ void cvt_f32_bf16_kernel(const float* __restrict__ in,
                                    __bf16* __restrict__ out, int n, int act) {
    int i = blockIdx.x * blockDim.x + threadIdx.x;
    if (i >= n) return;
    float v = in[i];
    if (act == 1) v = (v > 0.f) ? v : LEAKY * v;
    out[i] = (__bf16)v;
}

// ---- pack weight matrix W[K x Nout] (row-major f32) into B-fragment order ---
// Per 32(K) x 16(N) tile: lane l holds col n = ntile*16 + l%16, elements
// e<8 are K = ktile*32 + (l/16)*8 + e, e>=8 are K = ktile*32 + 16 + (l/16)*8 + (e-8).
__global__ void pack_w_kernel(const float* __restrict__ W,
                              __bf16* __restrict__ out, int K, int Nout) {
    int idx = blockIdx.x * blockDim.x + threadIdx.x;
    int total = K * Nout;
    if (idx >= total) return;
    int e      = idx & 15;
    int lane   = (idx >> 4) & 31;
    int tile   = idx >> 9;
    int ntiles = Nout >> 4;
    int ntile  = tile % ntiles;
    int ktile  = tile / ntiles;
    int n = ntile * 16 + (lane & 15);
    int k = ktile * 32 + ((e < 8) ? (lane >> 4) * 8 + e : 16 + (lane >> 4) * 8 + (e - 8));
    out[idx] = (__bf16)W[(size_t)k * Nout + n];
}

// ---- WMMA GEMM with LDS-staged A -------------------------------------------
// Grid.x = ceil(M/64); blockDim.x = 32*NTILES (compile-time).
// Wave w computes the 64x16 tile at column w*16. act: 0=none, 2=sigmoid.
template <int NTILES>
__global__ __launch_bounds__(32 * NTILES)
void gemm_wmma_bf16_kernel(const __bf16* __restrict__ A,
                           const __bf16* __restrict__ Bp,
                           float* __restrict__ C,
                           int M, int K, int act) {
    constexpr int THREADS = 32 * NTILES;
    constexpr int CHUNKS  = 256 / THREADS;    // 16B chunks per thread (1 or 2)
    constexpr int NOUT    = NTILES * 16;

    __shared__ __bf16 smA[2][64 * 32];        // 2 x 4KB double buffer

    const int lane  = threadIdx.x & 31;
    const int ntile = threadIdx.x >> 5;       // wave id == N tile
    const int mg    = blockIdx.x;             // 64-row strip index
    const int half  = lane >> 4;
    const int ktiles = K >> 5;

    // cooperative stage of A tile kt into LDS buffer `buf` (fully unrolled)
    auto stage = [&](int kt, int buf) {
#pragma unroll
        for (int it = 0; it < CHUNKS; ++it) {
            int c    = threadIdx.x + it * THREADS;  // 256 chunks: 4 per row
            int row  = c >> 2;
            int koff = (c & 3) * 8;
            int grow = mg * 64 + row;
            if (grow >= M) grow = M - 1;            // clamp (stores guarded)
            v8bf v = *(const v8bf*)(A + (size_t)grow * K + kt * 32 + koff);
            *(v8bf*)&smA[buf][row * 32 + koff] = v;
        }
    };

    v8f c0 = {}, c1 = {}, c2 = {}, c3 = {};

    stage(0, 0);
    __syncthreads();

    for (int kt = 0; kt < ktiles; ++kt) {
        int buf = kt & 1;

        // B fragment first (global load latency overlaps LDS traffic below)
        v16bf b = *(const v16bf*)(Bp + (((size_t)kt * NTILES + ntile) * 32 + lane) * 16);

        // prefetch next A tile into the other buffer
        if (kt + 1 < ktiles) stage(kt + 1, buf ^ 1);

        // fetch ALL four A fragments, then run the WMMA chain back-to-back
        const int rbase = (lane & 15) * 32 + half * 8;
        AFrag a0, a1, a2, a3;
        a0.h[0] = *(const v8bf*)&smA[buf][rbase + 0 * 512];
        a0.h[1] = *(const v8bf*)&smA[buf][rbase + 0 * 512 + 16];
        a1.h[0] = *(const v8bf*)&smA[buf][rbase + 1 * 512];
        a1.h[1] = *(const v8bf*)&smA[buf][rbase + 1 * 512 + 16];
        a2.h[0] = *(const v8bf*)&smA[buf][rbase + 2 * 512];
        a2.h[1] = *(const v8bf*)&smA[buf][rbase + 2 * 512 + 16];
        a3.h[0] = *(const v8bf*)&smA[buf][rbase + 3 * 512];
        a3.h[1] = *(const v8bf*)&smA[buf][rbase + 3 * 512 + 16];

        c0 = __builtin_amdgcn_wmma_f32_16x16x32_bf16(false, a0.v, false, b, (short)0, c0, false, false);
        c1 = __builtin_amdgcn_wmma_f32_16x16x32_bf16(false, a1.v, false, b, (short)0, c1, false, false);
        c2 = __builtin_amdgcn_wmma_f32_16x16x32_bf16(false, a2.v, false, b, (short)0, c2, false, false);
        c3 = __builtin_amdgcn_wmma_f32_16x16x32_bf16(false, a3.v, false, b, (short)0, c3, false, false);

        __syncthreads();
    }

    // Store D: VGPR r -> row = mt*16 + half*8 + r ; col = ntile*16 + lane%16
    const int col = ntile * 16 + (lane & 15);
    v8f acc[4] = {c0, c1, c2, c3};
    for (int i = 0; i < 4; ++i) {
        int rowbase = mg * 64 + i * 16 + half * 8;
        for (int r = 0; r < 8; ++r) {
            int row = rowbase + r;
            if (row < M) {
                float v = acc[i][r];
                if (act == 2) v = 1.0f / (1.0f + __expf(-v));
                C[(size_t)row * NOUT + col] = v;
            }
        }
    }
}

// ---- edge-parallel SpMM: out[dst] += h[src] * w  (128 feats, one wave/edge) -
__global__ __launch_bounds__(256)
void spmm128_kernel(const float* __restrict__ h,
                    const int* __restrict__ src, const int* __restrict__ dst,
                    const float* __restrict__ ew,
                    float* __restrict__ out, int E) {
    int edge = (blockIdx.x * blockDim.x + threadIdx.x) >> 5;
    int lane = threadIdx.x & 31;
    if (edge >= E) return;
    int s = src[edge];
    int d = dst[edge];
    float w = ew[edge];
    const float4 v = ((const float4*)(h + (size_t)s * NHID))[lane];
    float* op = out + (size_t)d * NHID + lane * 4;
    __hip_atomic_fetch_add(op + 0, v.x * w, __ATOMIC_RELAXED, __HIP_MEMORY_SCOPE_AGENT);
    __hip_atomic_fetch_add(op + 1, v.y * w, __ATOMIC_RELAXED, __HIP_MEMORY_SCOPE_AGENT);
    __hip_atomic_fetch_add(op + 2, v.z * w, __ATOMIC_RELAXED, __HIP_MEMORY_SCOPE_AGENT);
    __hip_atomic_fetch_add(op + 3, v.w * w, __ATOMIC_RELAXED, __HIP_MEMORY_SCOPE_AGENT);
}

// ---------------------------------------------------------------------------
extern "C" void kernel_launch(void* const* d_in, const int* in_sizes, int n_in,
                              void* d_out, int out_size, void* d_ws, size_t ws_size,
                              hipStream_t stream) {
    const float* x   = (const float*)d_in[0];
    const int*   es  = (const int*)  d_in[1];
    const int*   ed  = (const int*)  d_in[2];
    const float* ewt = (const float*)d_in[3];
    const float* W1  = (const float*)d_in[4];
    const float* W2  = (const float*)d_in[5];
    const float* W3  = (const float*)d_in[6];

    const int N = in_sizes[0] / NFEAT;   // 50000
    const int E = in_sizes[1];           // 800000

    // ---- workspace carving (256B aligned) ----
    char* ws = (char*)d_ws;
    size_t off = 0;
    auto carve = [&](size_t bytes) -> char* {
        char* p = ws + off;
        off = (off + bytes + 255) & ~(size_t)255;
        return p;
    };
    __bf16* Xb   = (__bf16*)carve((size_t)N * NFEAT * sizeof(__bf16));
    __bf16* W1b  = (__bf16*)carve((size_t)NFEAT * NHID * sizeof(__bf16));
    __bf16* W2b  = (__bf16*)carve((size_t)NHID * NHID * sizeof(__bf16));
    __bf16* W3b  = (__bf16*)carve((size_t)NHID * NCLASS * sizeof(__bf16));
    float*  H1   = (float*) carve((size_t)N * NHID * sizeof(float));
    float*  X1a  = (float*) carve((size_t)N * NHID * sizeof(float));
    __bf16* X1b  = (__bf16*)carve((size_t)N * NHID * sizeof(__bf16));
    float*  H2   = (float*) carve((size_t)N * NHID * sizeof(float));
    __bf16* X3b  = (__bf16*)carve((size_t)N * NHID * sizeof(__bf16));

    float* Y  = (float*)d_out;                      // [N, 64]
    float* X3 = (float*)d_out + (size_t)N * NCLASS; // [N, 128]

    const int TB = 256;
    const int mgroups = (N + 63) / 64;

    // 1) convert inputs to bf16
    {
        int n = N * NFEAT;
        cvt_f32_bf16_kernel<<<(n + TB - 1) / TB, TB, 0, stream>>>(x, Xb, n, 0);
    }
    pack_w_kernel<<<(NFEAT * NHID + TB - 1) / TB, TB, 0, stream>>>(W1, W1b, NFEAT, NHID);
    pack_w_kernel<<<(NHID * NHID + TB - 1) / TB, TB, 0, stream>>>(W2, W2b, NHID, NHID);
    pack_w_kernel<<<(NHID * NCLASS + TB - 1) / TB, TB, 0, stream>>>(W3, W3b, NHID, NCLASS);

    // 2) H1 = X @ W1   (8 waves/block, Nout = 128)
    gemm_wmma_bf16_kernel<NHID / 16><<<mgroups, 32 * (NHID / 16), 0, stream>>>(
        Xb, W1b, H1, N, NFEAT, 0);

    // 3) X1a = spmm(H1); then leaky + cvt -> X1b
    hipMemsetAsync(X1a, 0, (size_t)N * NHID * sizeof(float), stream);
    spmm128_kernel<<<(E * 32 + TB - 1) / TB, TB, 0, stream>>>(H1, es, ed, ewt, X1a, E);
    {
        int n = N * NHID;
        cvt_f32_bf16_kernel<<<(n + TB - 1) / TB, TB, 0, stream>>>(X1a, X1b, n, 1);
    }

    // 4) H2 = X1 @ W2
    gemm_wmma_bf16_kernel<NHID / 16><<<mgroups, 32 * (NHID / 16), 0, stream>>>(
        X1b, W2b, H2, N, NHID, 0);

    // 5) X3 = spmm(H2)  (directly into d_out's x3 slice)
    hipMemsetAsync(X3, 0, (size_t)N * NHID * sizeof(float), stream);
    spmm128_kernel<<<(E * 32 + TB - 1) / TB, TB, 0, stream>>>(H2, es, ed, ewt, X3, E);
    {
        int n = N * NHID;
        cvt_f32_bf16_kernel<<<(n + TB - 1) / TB, TB, 0, stream>>>(X3, X3b, n, 0);
    }

    // 6) Y = sigmoid(X3 @ W3)   (4 waves/block, Nout = 64)
    gemm_wmma_bf16_kernel<NCLASS / 16><<<mgroups, 32 * (NCLASS / 16), 0, stream>>>(
        X3b, W3b, Y, N, NHID, 2);
}
